// MyMambaBlock_81063212745233
// MI455X (gfx1250) — matrix operand order repaired and checked by reading, not compiled, not run on hardware
//
#include <hip/hip_runtime.h>
#include <hip/hip_bf16.h>

// Bidirectional Mamba block for MI455X (gfx1250, wave32).
// GEMMs: v_wmma_f32_16x16x32_bf16 (bf16 A/B, f32 accumulate), fragments built
// with 128-bit loads matching the CDNA5 WMMA VGPR layouts.
// Selective scan: 3-pass chunked linear-recurrence scan (16 chunks of 64) for
// 16x more parallelism on the serial critical path; chunk transition is
// exp(A * sum(delta)) so summaries are cheap.
// Workspace: float region (xin|z|xs|delta|yscan|Bm|Cm|Ssum|hq|hinit) followed
// by bf16 weight copies (W_in / W_x-padded-to-48 / W_out, both directions).

#define DM     64
#define ED     256
#define NS     16
#define LSEQ   1024
#define NB     16
#define DTR    4
#define WXPAD  48            // W_x rows padded 36 -> 48 (3 N-tiles)
#define NCHUNK 16
#define CLEN   (LSEQ / NCHUNK)   // 64

typedef __attribute__((ext_vector_type(16))) __bf16         v16bf;
typedef __attribute__((ext_vector_type(16))) unsigned short v16u;
typedef __attribute__((ext_vector_type(8)))  float          v8f;

__device__ __forceinline__ unsigned short f2bf(float f) {
    unsigned int u = __builtin_bit_cast(unsigned int, f);
    u += 0x7FFFu + ((u >> 16) & 1u);            // round-to-nearest-even
    return (unsigned short)(u >> 16);
}

__device__ __forceinline__ float silu(float v)     { return v / (1.0f + __expf(-v)); }
__device__ __forceinline__ float softplus(float v) { return (v > 20.0f) ? v : log1pf(__expf(v)); }

union frag_u { uint4 q[2]; v16u v; };

// A fragment (16x32 bf16, MxK). Lane m=lw&15; element i<8 holds K=kbase+i+8*hi,
// element i>=8 holds K=kbase+i+8+8*hi -> two contiguous 16-byte runs.
__device__ __forceinline__ v16bf a_frag(const unsigned short* row, int kbase, int hi) {
    frag_u u;
    u.q[0] = *reinterpret_cast<const uint4*>(row + kbase + 8 * hi);
    u.q[1] = *reinterpret_cast<const uint4*>(row + kbase + 16 + 8 * hi);
    return __builtin_bit_cast(v16bf, u.v);
}

// B fragment (32x16 bf16, KxN). Lane n=lw&15; element i<8 holds K=kbase+i+8*hi,
// element i>=8 holds K=kbase+i+8+8*hi -> two contiguous 16-byte runs.
__device__ __forceinline__ v16bf b_frag(const unsigned short* row, int kbase, int hi) {
    frag_u u;
    u.q[0] = *reinterpret_cast<const uint4*>(row + kbase + 8 * hi);
    u.q[1] = *reinterpret_cast<const uint4*>(row + kbase + 16 + 8 * hi);
    return __builtin_bit_cast(v16bf, u.v);
}

__device__ __forceinline__ v8f wmma_bf16(v16bf a, v16bf b, v8f c) {
    return __builtin_amdgcn_wmma_f32_16x16x32_bf16(false, a, false, b, (short)0, c, false, false);
}

// ---------------- Kernel 0: fp32 -> bf16 weight conversion (with zero pad) ----------------
__global__ __launch_bounds__(256) void k0_cvt_pad(
    const float* __restrict__ src, unsigned short* __restrict__ dst,
    int n_src, int n_dst)
{
    const int i = blockIdx.x * 256 + threadIdx.x;
    if (i < n_dst) dst[i] = (i < n_src) ? f2bf(src[i]) : (unsigned short)0;
}

// ---------------- Kernel 1: RMSNorm + W_in GEMM (64 -> 512) ----------------
// One block = 16 tokens of one direction (backward pre-flipped in time).
__global__ __launch_bounds__(256) void k1_norm_win(
    const float* __restrict__ x,
    const float* __restrict__ f_norm_w, const float* __restrict__ b_norm_w,
    const unsigned short* __restrict__ f_W_in_bf, const unsigned short* __restrict__ b_W_in_bf,
    float* __restrict__ xin, float* __restrict__ zbuf)
{
    __shared__ alignas(16) unsigned short xn_lds[16][DM];
    __shared__ float red[16][16];
    __shared__ float scale[16];

    const int tid  = threadIdx.x;
    const int tilesPerDir = (NB * LSEQ) / 16;            // 1024
    const int dir  = blockIdx.x / tilesPerDir;
    const int rem  = blockIdx.x % tilesPerDir;
    const int bIdx = (rem * 16) / LSEQ;
    const int lBase = (rem * 16) % LSEQ;

    const float* norm_w = dir ? b_norm_w : f_norm_w;
    const unsigned short* W_in = dir ? b_W_in_bf : f_W_in_bf;

    // RMSNorm: token r (0..15), thread group c (0..15) covers 4 dims each.
    const int r = tid >> 4;
    const int c = tid & 15;
    const int l    = lBase + r;
    const int lsrc = dir ? (LSEQ - 1 - l) : l;
    const float* xp = x + ((size_t)bIdx * LSEQ + lsrc) * DM + c * 4;
    float xv[4];
    float ss = 0.0f;
#pragma unroll
    for (int i = 0; i < 4; ++i) { xv[i] = xp[i]; ss += xv[i] * xv[i]; }
    red[r][c] = ss;
    __syncthreads();
    if (tid < 16) {
        float s = 0.0f;
#pragma unroll
        for (int i = 0; i < 16; ++i) s += red[tid][i];
        scale[tid] = rsqrtf(s / (float)DM + 1e-5f);
    }
    __syncthreads();
    const float sc = scale[r];
#pragma unroll
    for (int i = 0; i < 4; ++i)
        xn_lds[r][c * 4 + i] = f2bf(xv[i] * sc * norm_w[c * 4 + i]);
    __syncthreads();

    // GEMM: 16 tokens x 512 features, K = 64. 8 waves x 4 N-tiles each.
    const int wave = tid >> 5;
    const int lw   = tid & 31;
    const int hi   = lw >> 4;
    const unsigned short* arow = &xn_lds[lw & 15][0];
    const size_t tokOff = ((size_t)(dir * NB + bIdx) * LSEQ + lBase) * ED;

#pragma unroll
    for (int nt = 0; nt < 4; ++nt) {
        const int fb = (wave * 4 + nt) * 16;             // tile base (wave-uniform)
        const int f  = fb + (lw & 15);                   // output feature column
        const unsigned short* wrow = W_in + (size_t)f * DM;
        v8f acc = {};
#pragma unroll
        for (int ks = 0; ks < 2; ++ks)
            acc = wmma_bf16(a_frag(arow, ks * 32, hi), b_frag(wrow, ks * 32, hi), acc);

        float* dst = (fb < ED) ? xin : zbuf;             // wave-uniform select
        const int fc = (fb < ED) ? f : (f - ED);
#pragma unroll
        for (int rr = 0; rr < 8; ++rr) {
            const int mm = rr + 8 * hi;                  // token row
            dst[tokOff + (size_t)mm * ED + fc] = acc[rr];
        }
    }
}

// ---------------- Kernel 2: causal depthwise conv + SiLU ----------------
__global__ __launch_bounds__(256) void k2_conv_silu(
    const float* __restrict__ xin,
    const float* __restrict__ f_conv_w, const float* __restrict__ f_conv_b,
    const float* __restrict__ b_conv_w, const float* __restrict__ b_conv_b,
    float* __restrict__ xs)
{
    const size_t idx = (size_t)blockIdx.x * blockDim.x + threadIdx.x;
    const int e    = (int)(idx & (ED - 1));
    const int l    = (int)((idx >> 8) & (LSEQ - 1));
    const int bIdx = (int)((idx >> 18) & (NB - 1));
    const int dir  = (int)(idx >> 22);

    const float* cw = dir ? b_conv_w : f_conv_w;
    const float* cb = dir ? b_conv_b : f_conv_b;

    const size_t base = ((size_t)(dir * NB + bIdx) * LSEQ) * ED + e;
    float s = cb[e];
#pragma unroll
    for (int k = 0; k < 4; ++k) {
        const int ll = l - 3 + k;
        if (ll >= 0) s += cw[e * 4 + k] * xin[base + (size_t)ll * ED];
    }
    xs[base + (size_t)l * ED] = silu(s);
}

// ---------------- Kernel 3: W_x GEMM (256 -> 36, padded 48) + delta ----------------
// One block = 16 tokens, 4 waves; waves 0..2 compute the 3 N-tiles.
__global__ __launch_bounds__(128) void k3_wx_delta(
    const float* __restrict__ xs,
    const unsigned short* __restrict__ f_W_x_bf, const unsigned short* __restrict__ b_W_x_bf,
    const float* __restrict__ f_W_dt, const float* __restrict__ f_b_dt,
    const float* __restrict__ b_W_dt, const float* __restrict__ b_b_dt,
    float* __restrict__ Bm, float* __restrict__ Cm, float* __restrict__ delta)
{
    __shared__ alignas(16) unsigned short xs_lds[16][ED];
    __shared__ float dt_lds[16][DTR];

    const int tid = threadIdx.x;
    const int tilesPerDir = (NB * LSEQ) / 16;
    const int dir  = blockIdx.x / tilesPerDir;
    const int rem  = blockIdx.x % tilesPerDir;
    const int bIdx = (rem * 16) / LSEQ;
    const int lBase = (rem * 16) % LSEQ;

    const unsigned short* W_x = dir ? b_W_x_bf : f_W_x_bf;
    const float* W_dt = dir ? b_W_dt : f_W_dt;
    const float* b_dt = dir ? b_b_dt : f_b_dt;

    const size_t base = ((size_t)(dir * NB + bIdx) * LSEQ + lBase) * ED;
    for (int i = tid; i < 16 * ED; i += 128)
        xs_lds[i >> 8][i & (ED - 1)] = f2bf(xs[base + i]);
    __syncthreads();

    const int wave = tid >> 5;
    const int lw   = tid & 31;
    const int hi   = lw >> 4;
    if (wave < 3) {
        const int f = wave * 16 + (lw & 15);             // padded dbl row (0..47)
        const unsigned short* wrow = W_x + (size_t)f * ED;
        const unsigned short* arow = &xs_lds[lw & 15][0];
        v8f acc = {};
#pragma unroll
        for (int ks = 0; ks < 8; ++ks)
            acc = wmma_bf16(a_frag(arow, ks * 32, hi), b_frag(wrow, ks * 32, hi), acc);

        const size_t bcTok = ((size_t)(dir * NB + bIdx) * LSEQ + lBase);
#pragma unroll
        for (int rr = 0; rr < 8; ++rr) {
            const int mm = rr + 8 * hi;
            if (f < DTR)               dt_lds[mm][f] = acc[rr];
            else if (f < DTR + NS)     Bm[(bcTok + mm) * NS + (f - DTR)]      = acc[rr];
            else if (f < DTR + 2 * NS) Cm[(bcTok + mm) * NS + (f - DTR - NS)] = acc[rr];
        }
    }
    __syncthreads();

    // delta = softplus(dt @ W_dt^T + b_dt), 16 tokens x 256 channels
    for (int i = tid; i < 16 * ED; i += 128) {
        const int row = i >> 8;
        const int e   = i & (ED - 1);
        float s = b_dt[e];
#pragma unroll
        for (int rr = 0; rr < DTR; ++rr) s += dt_lds[row][rr] * W_dt[e * DTR + rr];
        delta[base + i] = softplus(s);
    }
}

// ---------------- Kernel 4a: per-chunk local scan + chunk summary ----------------
// Grid: (2*NB*NCHUNK) blocks; thread e owns channel e, 16 states in registers.
// Summary: S = sum(delta) over chunk (chunk transition = exp(A*S)), q = local
// end-state with h0 = 0.
__global__ __launch_bounds__(256) void k4a_chunk_scan(
    const float* __restrict__ xs, const float* __restrict__ delta,
    const float* __restrict__ Bm,
    const float* __restrict__ f_A_log, const float* __restrict__ b_A_log,
    float* __restrict__ Ssum, float* __restrict__ hq)
{
    __shared__ float sB[CLEN][NS];

    const int g   = blockIdx.x / NCHUNK;      // dir*NB + b
    const int ch  = blockIdx.x % NCHUNK;
    const int dir = g >> 4;
    const int e   = threadIdx.x;
    const float* A_log = dir ? b_A_log : f_A_log;

    float A[NS], h[NS];
#pragma unroll
    for (int n = 0; n < NS; ++n) { A[n] = -__expf(A_log[e * NS + n]); h[n] = 0.0f; }

    const size_t bc = ((size_t)g * LSEQ + ch * CLEN) * NS;
    for (int i = threadIdx.x; i < CLEN * NS; i += 256) sB[i >> 4][i & 15] = Bm[bc + i];
    __syncthreads();

    const size_t tb = ((size_t)g * LSEQ + ch * CLEN) * ED + e;
    float S = 0.0f;
    for (int t = 0; t < CLEN; ++t) {
        __builtin_prefetch(delta + tb + (size_t)(t + 16) * ED, 0, 1);
        const float dt = delta[tb + (size_t)t * ED];
        const float xt = xs[tb + (size_t)t * ED];
        const float dx = dt * xt;
        S += dt;
#pragma unroll
        for (int n = 0; n < NS; ++n)
            h[n] = h[n] * __expf(dt * A[n]) + dx * sB[t][n];
    }

    Ssum[((size_t)g * NCHUNK + ch) * ED + e] = S;
    float* hp = hq + (((size_t)g * NCHUNK + ch) * ED + e) * NS;
#pragma unroll
    for (int n = 0; n < NS; ++n) hp[n] = h[n];
}

// ---------------- Kernel 4b: combine chunk summaries (only NCHUNK serial steps) ----------------
__global__ __launch_bounds__(256) void k4b_combine(
    const float* __restrict__ Ssum, const float* __restrict__ hq,
    const float* __restrict__ f_A_log, const float* __restrict__ b_A_log,
    float* __restrict__ hinit)
{
    const int g   = blockIdx.x;               // dir*NB + b
    const int dir = g >> 4;
    const int e   = threadIdx.x;
    const float* A_log = dir ? b_A_log : f_A_log;

    float A[NS], h[NS];
#pragma unroll
    for (int n = 0; n < NS; ++n) { A[n] = -__expf(A_log[e * NS + n]); h[n] = 0.0f; }

    for (int c = 0; c < NCHUNK; ++c) {
        const size_t idx = ((size_t)g * NCHUNK + c) * ED + e;
        float* hi = hinit + idx * NS;
#pragma unroll
        for (int n = 0; n < NS; ++n) hi[n] = h[n];

        const float S = Ssum[idx];
        const float* q = hq + idx * NS;
#pragma unroll
        for (int n = 0; n < NS; ++n) h[n] = h[n] * __expf(A[n] * S) + q[n];
    }
}

// ---------------- Kernel 4c: final chunk scan with true initial states + outputs ----------------
__global__ __launch_bounds__(256) void k4c_final_scan(
    const float* __restrict__ xs, const float* __restrict__ delta,
    const float* __restrict__ Bm, const float* __restrict__ Cm,
    const float* __restrict__ hinit,
    const float* __restrict__ f_A_log, const float* __restrict__ f_D,
    const float* __restrict__ b_A_log, const float* __restrict__ b_D,
    float* __restrict__ yscan)
{
    __shared__ float sB[CLEN][NS];
    __shared__ float sC[CLEN][NS];

    const int g   = blockIdx.x / NCHUNK;
    const int ch  = blockIdx.x % NCHUNK;
    const int dir = g >> 4;
    const int e   = threadIdx.x;
    const float* A_log = dir ? b_A_log : f_A_log;
    const float* Dp    = dir ? b_D     : f_D;

    float A[NS], h[NS];
    const float* hi = hinit + (((size_t)g * NCHUNK + ch) * ED + e) * NS;
#pragma unroll
    for (int n = 0; n < NS; ++n) { A[n] = -__expf(A_log[e * NS + n]); h[n] = hi[n]; }
    const float Dd = Dp[e];

    const size_t bc = ((size_t)g * LSEQ + ch * CLEN) * NS;
    for (int i = threadIdx.x; i < CLEN * NS; i += 256) {
        sB[i >> 4][i & 15] = Bm[bc + i];
        sC[i >> 4][i & 15] = Cm[bc + i];
    }
    __syncthreads();

    const size_t tb = ((size_t)g * LSEQ + ch * CLEN) * ED + e;
    for (int t = 0; t < CLEN; ++t) {
        __builtin_prefetch(delta + tb + (size_t)(t + 16) * ED, 0, 1);
        __builtin_prefetch(xs    + tb + (size_t)(t + 16) * ED, 0, 1);
        const float dt = delta[tb + (size_t)t * ED];
        const float xt = xs[tb + (size_t)t * ED];
        const float dx = dt * xt;
        float y = 0.0f;
#pragma unroll
        for (int n = 0; n < NS; ++n) {
            h[n] = h[n] * __expf(dt * A[n]) + dx * sB[t][n];
            y += h[n] * sC[t][n];
        }
        yscan[tb + (size_t)t * ED] = y + xt * Dd;
    }
}

// ---------------- Kernel 5: gate + W_out GEMM (256 -> 64) + residual ----------------
__global__ __launch_bounds__(128) void k5_out(
    const float* __restrict__ x,
    const float* __restrict__ yscan, const float* __restrict__ zbuf,
    const unsigned short* __restrict__ f_W_out_bf, const unsigned short* __restrict__ b_W_out_bf,
    float* __restrict__ out)
{
    __shared__ alignas(16) unsigned short y_lds[16][ED];

    const int tid = threadIdx.x;
    const int tilesPerDir = (NB * LSEQ) / 16;
    const int dir  = blockIdx.x / tilesPerDir;
    const int rem  = blockIdx.x % tilesPerDir;
    const int bIdx = (rem * 16) / LSEQ;
    const int lBase = (rem * 16) % LSEQ;

    const unsigned short* W_out = dir ? b_W_out_bf : f_W_out_bf;

    const size_t base = ((size_t)(dir * NB + bIdx) * LSEQ + lBase) * ED;
    for (int i = tid; i < 16 * ED; i += 128) {
        const float zv = zbuf[base + i];
        y_lds[i >> 8][i & (ED - 1)] = f2bf(yscan[base + i] * silu(zv));
    }
    __syncthreads();

    const int wave = tid >> 5;                            // 0..3 -> 4 N-tiles of 16
    const int lw   = tid & 31;
    const int hi   = lw >> 4;
    const int dm   = wave * 16 + (lw & 15);               // output model dim 0..63
    const unsigned short* arow = &y_lds[lw & 15][0];
    const unsigned short* wrow = W_out + (size_t)dm * ED;

    v8f acc = {};
#pragma unroll
    for (int ks = 0; ks < 8; ++ks)
        acc = wmma_bf16(a_frag(arow, ks * 32, hi), b_frag(wrow, ks * 32, hi), acc);

#pragma unroll
    for (int rr = 0; rr < 8; ++rr) {
        const int mm = rr + 8 * hi;
        const int l  = lBase + mm;
        const int lo = dir ? (LSEQ - 1 - l) : l;          // un-flip backward dir
        const float resid = x[((size_t)bIdx * LSEQ + lo) * DM + dm];
        out[((size_t)bIdx * LSEQ + lo) * (2 * DM) + dir * DM + dm] = acc[rr] + resid;
    }
}

extern "C" void kernel_launch(void* const* d_in, const int* in_sizes, int n_in,
                              void* d_out, int out_size, void* d_ws, size_t ws_size,
                              hipStream_t stream) {
    (void)in_sizes; (void)n_in; (void)out_size; (void)ws_size;

    const float* x        = (const float*)d_in[0];
    const float* f_norm_w = (const float*)d_in[1];
    const float* f_W_in   = (const float*)d_in[2];
    const float* f_conv_w = (const float*)d_in[3];
    const float* f_conv_b = (const float*)d_in[4];
    const float* f_W_x    = (const float*)d_in[5];
    const float* f_W_dt   = (const float*)d_in[6];
    const float* f_b_dt   = (const float*)d_in[7];
    const float* f_A_log  = (const float*)d_in[8];
    const float* f_D      = (const float*)d_in[9];
    const float* f_W_out  = (const float*)d_in[10];
    const float* b_norm_w = (const float*)d_in[11];
    const float* b_W_in   = (const float*)d_in[12];
    const float* b_conv_w = (const float*)d_in[13];
    const float* b_conv_b = (const float*)d_in[14];
    const float* b_W_x    = (const float*)d_in[15];
    const float* b_W_dt   = (const float*)d_in[16];
    const float* b_b_dt   = (const float*)d_in[17];
    const float* b_A_log  = (const float*)d_in[18];
    const float* b_D      = (const float*)d_in[19];
    const float* b_W_out  = (const float*)d_in[20];

    float* ws = (float*)d_ws;
    const size_t TOK = (size_t)2 * NB * LSEQ;             // tokens across both dirs
    const size_t NCH = (size_t)2 * NB * NCHUNK;           // chunks across both dirs
    float* xin   = ws;
    float* zbuf  = xin   + TOK * ED;
    float* xs    = zbuf  + TOK * ED;
    float* delta = xs    + TOK * ED;
    float* ysc   = delta + TOK * ED;
    float* Bm    = ysc   + TOK * ED;
    float* Cm    = Bm    + TOK * NS;
    float* Ssum  = Cm    + TOK * NS;                      // [NCH][ED]
    float* hq    = Ssum  + NCH * ED;                      // [NCH][ED][NS]
    float* hinit = hq    + NCH * ED * NS;                 // [NCH][ED][NS]

    // bf16 weight copies after the float region
    unsigned short* wb      = (unsigned short*)(hinit + NCH * ED * NS);
    unsigned short* fWin_bf = wb;                                  // 512*64
    unsigned short* bWin_bf = fWin_bf + 2 * ED * DM;
    unsigned short* fWx_bf  = bWin_bf + 2 * ED * DM;               // 48*256 (padded)
    unsigned short* bWx_bf  = fWx_bf  + WXPAD * ED;
    unsigned short* fWo_bf  = bWx_bf  + WXPAD * ED;                // 64*256
    unsigned short* bWo_bf  = fWo_bf  + DM * ED;

    const int nWin = 2 * ED * DM;                                  // 32768
    const int nWx  = (DTR + 2 * NS) * ED;                          // 9216
    const int nWxP = WXPAD * ED;                                   // 12288
    const int nWo  = DM * ED;                                      // 16384

    k0_cvt_pad<<<(nWin + 255) / 256, 256, 0, stream>>>(f_W_in,  fWin_bf, nWin, nWin);
    k0_cvt_pad<<<(nWin + 255) / 256, 256, 0, stream>>>(b_W_in,  bWin_bf, nWin, nWin);
    k0_cvt_pad<<<(nWxP + 255) / 256, 256, 0, stream>>>(f_W_x,   fWx_bf,  nWx,  nWxP);
    k0_cvt_pad<<<(nWxP + 255) / 256, 256, 0, stream>>>(b_W_x,   bWx_bf,  nWx,  nWxP);
    k0_cvt_pad<<<(nWo + 255) / 256, 256, 0, stream>>>(f_W_out, fWo_bf,  nWo,  nWo);
    k0_cvt_pad<<<(nWo + 255) / 256, 256, 0, stream>>>(b_W_out, bWo_bf,  nWo,  nWo);

    const int tileBlocks = (int)(TOK / 16);               // 2048

    k1_norm_win<<<tileBlocks, 256, 0, stream>>>(x, f_norm_w, b_norm_w, fWin_bf, bWin_bf, xin, zbuf);
    k2_conv_silu<<<(int)(TOK * ED / 256), 256, 0, stream>>>(xin, f_conv_w, f_conv_b, b_conv_w, b_conv_b, xs);
    k3_wx_delta<<<tileBlocks, 128, 0, stream>>>(xs, fWx_bf, bWx_bf, f_W_dt, f_b_dt, b_W_dt, b_b_dt, Bm, Cm, delta);

    k4a_chunk_scan<<<(int)NCH, 256, 0, stream>>>(xs, delta, Bm, f_A_log, b_A_log, Ssum, hq);
    k4b_combine<<<2 * NB, 256, 0, stream>>>(Ssum, hq, f_A_log, b_A_log, hinit);
    k4c_final_scan<<<(int)NCH, 256, 0, stream>>>(xs, delta, Bm, Cm, hinit, f_A_log, f_D, b_A_log, b_D, ysc);

    k5_out<<<tileBlocks, 128, 0, stream>>>(x, ysc, zbuf, fWo_bf, bWo_bf, (float*)d_out);
}
